// EncoderBlock_609885356077
// MI455X (gfx1250) — matrix operand order repaired and checked by reading, not compiled, not run on hardware
//
#include <hip/hip_runtime.h>

// ---------------------------------------------------------------------------
// CDNA5 WMMA (wave32): D(16x16 f32) = A(16x32 f16) x B(32x16 f16) + C
// ---------------------------------------------------------------------------
typedef __attribute__((ext_vector_type(16))) _Float16 v16h;
typedef __attribute__((ext_vector_type(8)))  float    v8f;
typedef __attribute__((ext_vector_type(4)))  unsigned int v4u;
typedef __attribute__((ext_vector_type(8)))  int      v8i;
typedef __attribute__((ext_vector_type(4)))  int      v4i;

#define WMMA_F16(a, b, c) \
  __builtin_amdgcn_wmma_f32_16x16x32_f16(false, (a), false, (b), (short)0, (c), false, false)

#if defined(__has_builtin)
#  if __has_builtin(__builtin_amdgcn_tensor_load_to_lds)
#    define ATHENA_HAVE_TDM 1
#  endif
#  if __has_builtin(__builtin_amdgcn_s_wait_tensorcnt)
#    define ATHENA_HAVE_TENSORCNT 1
#  endif
#endif
#ifndef ATHENA_HAVE_TDM
#  define ATHENA_HAVE_TDM 0
#endif
#ifndef ATHENA_HAVE_TENSORCNT
#  define ATHENA_HAVE_TENSORCNT 0
#endif

#if ATHENA_HAVE_TDM
// 2D tile load via Tensor Data Mover (ISA 08_async_tensor, D# groups 0/1).
// Loads tile_h rows of tile_w 2-byte elements (row stride row_stride_elems)
// from gaddr into LDS at lds_addr, row-major contiguous.
__device__ __forceinline__ void tdm_load_2d_to_lds(unsigned lds_addr, const void* gaddr,
                                                   unsigned tile_w, unsigned tile_h,
                                                   unsigned row_stride_elems) {
  const unsigned long long ga = (unsigned long long)gaddr;
  v4u g0;
  g0[0] = 1u;                                                   // count=1, user D#
  g0[1] = lds_addr;                                             // LDS byte address
  g0[2] = (unsigned)ga;                                         // global_addr[31:0]
  g0[3] = (unsigned)((ga >> 32) & 0x01FFFFFFu) | (2u << 30);    // addr[56:32], type=2
  v8i g1;
  g1[0] = (int)(1u << 16);               // wg_mask=0, data_size=1 (2 bytes)
  g1[1] = (int)(tile_w << 16);           // tensor_dim0[15:0]
  g1[2] = (int)(tile_h << 16);           // tensor_dim0[31:16]=0, tensor_dim1[15:0]
  g1[3] = (int)(tile_w << 16);           // tensor_dim1[31:16]=0, tile_dim0
  g1[4] = (int)tile_h;                   // tile_dim1, tile_dim2=0
  g1[5] = (int)row_stride_elems;         // tensor_dim0_stride[31:0]
  g1[6] = 0;                             // stride[47:32]=0, dim1_stride[15:0]=0
  g1[7] = 0;
  v4i z4 = {0, 0, 0, 0};
#if __clang_major__ >= 23
  v8i z8 = {0, 0, 0, 0, 0, 0, 0, 0};
  __builtin_amdgcn_tensor_load_to_lds(g0, g1, z4, z4, z8, 0);
#else
  __builtin_amdgcn_tensor_load_to_lds(g0, g1, z4, z4, 0);
#endif
}
#endif

__device__ __forceinline__ void tdm_wait0() {
#if ATHENA_HAVE_TENSORCNT
  __builtin_amdgcn_s_wait_tensorcnt(0);
#else
  asm volatile("s_wait_tensorcnt 0" ::: "memory");
#endif
}

// A-matrix (16x32 f16) per-lane K index, ISA 7.12.2.
__device__ __forceinline__ int a_kidx(int e, int g) {
  int vg = e >> 1, pos = e & 1;
  return (vg < 4) ? (8 * g + 2 * vg + pos) : (16 + 8 * g + 2 * (vg - 4) + pos);
}

__device__ __forceinline__ float gelu_exact(float v) {
  return 0.5f * v * (1.0f + erff(v * 0.70710678118654752f));
}

// ---------------------------------------------------------------------------
// fp32 -> f16 cast (weights)
// ---------------------------------------------------------------------------
__global__ void cast_f32_f16(const float* __restrict__ in, _Float16* __restrict__ out, int n) {
  int i = blockIdx.x * blockDim.x + threadIdx.x;
  int stride = gridDim.x * blockDim.x;
  for (; i < n; i += stride) out[i] = (_Float16)in[i];
}

// ---------------------------------------------------------------------------
// LayerNorm over D=768, one workgroup (8 waves) per row. OUT16: emit f16.
// ---------------------------------------------------------------------------
template <bool OUT16>
__global__ __launch_bounds__(256) void ln_kernel(const float* __restrict__ x,
                                                 const float* __restrict__ gam,
                                                 const float* __restrict__ bet,
                                                 void* __restrict__ out) {
  const int row = blockIdx.x;
  const float* xr = x + (size_t)row * 768;
  float s = 0.f, ss = 0.f;
  for (int i = threadIdx.x; i < 768; i += 256) {
    float v = xr[i];
    s += v; ss += v * v;
  }
#pragma unroll
  for (int off = 1; off < 32; off <<= 1) {
    s  += __shfl_xor(s, off, 32);
    ss += __shfl_xor(ss, off, 32);
  }
  __shared__ float sm[8], sm2[8];
  const int w = threadIdx.x >> 5, lane = threadIdx.x & 31;
  if (lane == 0) { sm[w] = s; sm2[w] = ss; }
  __syncthreads();
  float ts = 0.f, tss = 0.f;
#pragma unroll
  for (int i = 0; i < 8; ++i) { ts += sm[i]; tss += sm2[i]; }
  const float mu  = ts * (1.0f / 768.0f);
  const float var = tss * (1.0f / 768.0f) - mu * mu;
  const float inv = rsqrtf(var + 1e-5f);
  for (int i = threadIdx.x; i < 768; i += 256) {
    float v = (xr[i] - mu) * inv * gam[i] + bet[i];
    if (OUT16) ((_Float16*)out)[(size_t)row * 768 + i] = (_Float16)v;
    else       ((float*)out)[(size_t)row * 768 + i]    = v;
  }
}

// ---------------------------------------------------------------------------
// Tiled WMMA GEMM: C[M,Nc] = A16[M,K] @ B16[K,Nc] (+bias +GELU +residual)
// 256 threads = 8 waves (4x2); block tile 128x128; wave tile 32x64
// (2x4 accumulators -> 8 WMMA per wave per 32-wide K step).
// Double-buffered LDS, one barrier per K step, register-staged global loads.
// A stored in LDS in WMMA-fragment order; B stored transposed -> all
// fragment loads are ds_load_b128 pairs.
// ---------------------------------------------------------------------------
template <bool BIAS, bool GELU, bool RES, bool OUT16>
__global__ __launch_bounds__(256) void gemm_wmma(const _Float16* __restrict__ A,
                                                 const _Float16* __restrict__ Bm,
                                                 const float* __restrict__ bias,
                                                 const float* __restrict__ res,
                                                 void* __restrict__ Cout,
                                                 int M, int K, int Nc) {
  // Afrag[buf][rowtile(8)][lane(32)][e(16)] : per-lane fragment contiguous (32B)
  __shared__ __align__(32) _Float16 Afrag[2][8][32][16];
  // Bt[buf][col(128)][k(32)] + pad to 40 halves (80B rows) against bank conflicts
  __shared__ __align__(32) _Float16 Bt[2][128][40];

  const int bm = blockIdx.y * 128, bn = blockIdx.x * 128;
  const int t = threadIdx.x, lane = t & 31, w = t >> 5;
  const int wm = (w & 3) * 32, wn = (w >> 2) * 64;  // wave tile origin
  const int r = lane & 15, g = lane >> 4;

  // staging coords: each thread moves 32B of A and 32B of B per K step
  const int arow = t >> 1, ahalf = t & 1;         // A: row, K-half (0..15 vs 16..31)
  const int bkrow = t >> 3, bcol = (t & 7) * 16;  // B: k row, 16-column chunk

  uint4 a0v, a1v, b0v, b1v;

  auto load_slab = [&](int k0) {
    const uint4* ap = (const uint4*)(A + (size_t)(bm + arow) * K + k0 + ahalf * 16);
    a0v = ap[0]; a1v = ap[1];
    const uint4* bp = (const uint4*)(Bm + (size_t)(k0 + bkrow) * Nc + bn + bcol);
    b0v = bp[0]; b1v = bp[1];
  };
  auto store_slab = [&](int buf) {
    // A chunks of 8 contiguous K map to (lane group, e-offset): chunk c -> g=c>>1, eoff=8*(c&1)
    const int eoff = ahalf * 16;
    *(uint4*)&Afrag[buf][arow >> 4][eoff + (arow & 15)][0] = a0v;  // g=ahalf lane
    *(uint4*)&Afrag[buf][arow >> 4][eoff + (arow & 15)][8] = a1v;  // g=ahalf lane
    union { uint4 u; _Float16 h[8]; } u0, u1;
    u0.u = b0v; u1.u = b1v;
#pragma unroll
    for (int jj = 0; jj < 8; ++jj) {
      Bt[buf][bcol + jj][bkrow]     = u0.h[jj];
      Bt[buf][bcol + 8 + jj][bkrow] = u1.h[jj];
    }
  };

  v8f acc[2][4] = {};
  const int rt0 = (w & 3) * 2;  // this wave's two 16-row fragment tiles

  load_slab(0);
  store_slab(0);
  __syncthreads();

  int buf = 0;
  for (int k0 = 0; k0 < K; k0 += 32) {
    const bool more = (k0 + 32 < K);
    if (more) load_slab(k0 + 32);

    const v16h af0 = *(const v16h*)&Afrag[buf][rt0][lane][0];
    const v16h af1 = *(const v16h*)&Afrag[buf][rt0 + 1][lane][0];
#pragma unroll
    for (int s = 0; s < 4; ++s) {
      const v16h bf = *(const v16h*)&Bt[buf][wn + s * 16 + r][16 * g];
      acc[0][s] = WMMA_F16(af0, bf, acc[0][s]);
      acc[1][s] = WMMA_F16(af1, bf, acc[1][s]);
    }
    if (more) store_slab(buf ^ 1);
    __syncthreads();
    buf ^= 1;
  }

#pragma unroll
  for (int rt = 0; rt < 2; ++rt)
#pragma unroll
    for (int s = 0; s < 4; ++s)
#pragma unroll
      for (int p = 0; p < 8; ++p) {
        const int row = bm + wm + rt * 16 + p + 8 * g;
        const int col = bn + wn + s * 16 + r;
        float v = acc[rt][s][p];
        if (BIAS) v += bias[col];
        if (GELU) v = gelu_exact(v);
        if (RES)  v += res[(size_t)row * Nc + col];
        if (OUT16) ((_Float16*)Cout)[(size_t)row * Nc + col] = (_Float16)v;
        else       ((float*)Cout)[(size_t)row * Nc + col]    = v;
      }
}

// ---------------------------------------------------------------------------
// Flash-style attention. Grid (B*H, N/64), 4 waves/block, 16 queries/wave.
// K tile (32 keys x 96) staged to LDS by the Tensor Data Mover (row-major,
// exactly the B-fragment-friendly layout); V tile staged manually transposed.
// qkv layout: [(b*1024+n)*2304 + {0|768|1536} + h*96 + dh], f16.
// Output in (B,H,N,Dh) order == reference's head-major reshape to (B,N,D).
// ---------------------------------------------------------------------------
__global__ __launch_bounds__(128) void attn_kernel(const _Float16* __restrict__ qkv,
                                                   _Float16* __restrict__ out) {
  const int bh = blockIdx.x;
  const int b = bh >> 3, h = bh & 7;
  const int t = threadIdx.x, wv = t >> 5, lane = t & 31;
  const int r = lane & 15, g = lane >> 4;
  const int qt = blockIdx.y * 64 + wv * 16;

  __shared__ __align__(32) _Float16 Klds[32][96];   // keys x dh (row-major)
  __shared__ __align__(32) _Float16 Vt[96][32];     // dh x keys (transposed)
  __shared__ __align__(32) _Float16 Plds[4][16][32];

  const size_t RS = 2304;
  const size_t base = (size_t)b * 1024 * RS + (size_t)h * 96;

  // Q: 3 A-fragments (K = dh 0..95), loaded once per wave
  v16h qf[3];
#pragma unroll
  for (int kk = 0; kk < 3; ++kk) {
    const _Float16* qp = qkv + base + (size_t)(qt + r) * RS + kk * 32;
#pragma unroll
    for (int e = 0; e < 16; ++e) qf[kk][e] = qp[16 * g + e];
  }

  v8f o[6];
#pragma unroll
  for (int n = 0; n < 6; ++n) o[n] = (v8f){};
  float mrow[8], lrow[8];
#pragma unroll
  for (int p = 0; p < 8; ++p) { mrow[p] = -3.0e38f; lrow[p] = 0.0f; }

  const float scale = 0.1020620726f;  // 96^-0.5

  for (int j = 0; j < 1024; j += 32) {
#if ATHENA_HAVE_TDM
    // ---- K tile via TDM: 32 rows x 96 halves, row stride 2304 -> Klds ----
    if (wv == 0) {
      tdm_load_2d_to_lds((unsigned)(size_t)&Klds[0][0],
                         qkv + base + (size_t)j * RS + 768,
                         96u, 32u, 2304u);
      tdm_wait0();
    }
#endif
    // ---- V tile (transposed) staged manually; K manual if no TDM ----
#pragma unroll
    for (int i = 0; i < 3; ++i) {
      const int idx = t + 128 * i;          // 0..383 uint4 tiles
      const int row = idx / 12, c = (idx % 12) * 8;
#if !ATHENA_HAVE_TDM
      const uint4 kv = *(const uint4*)(qkv + base + (size_t)(j + row) * RS + 768 + c);
      *(uint4*)&Klds[row][c] = kv;
#endif
      union { uint4 u; _Float16 hh[8]; } vv;
      vv.u = *(const uint4*)(qkv + base + (size_t)(j + row) * RS + 1536 + c);
#pragma unroll
      for (int jj = 0; jj < 8; ++jj) Vt[c + jj][row] = vv.hh[jj];
    }
    __syncthreads();

    // ---- S = Q @ K^T (two 16-key fragments) ----
    v8f s0 = {}, s1 = {};
#pragma unroll
    for (int kk = 0; kk < 3; ++kk) {
      const v16h kf0 = *(const v16h*)&Klds[r][kk * 32 + 16 * g];
      const v16h kf1 = *(const v16h*)&Klds[16 + r][kk * 32 + 16 * g];
      s0 = WMMA_F16(qf[kk], kf0, s0);
      s1 = WMMA_F16(qf[kk], kf1, s1);
    }

    // ---- online softmax (rows spread over p, cols over 16 lanes) ----
    float alpha[8];
#pragma unroll
    for (int p = 0; p < 8; ++p) {
      float a0 = s0[p] * scale, a1 = s1[p] * scale;
      float mx = fmaxf(a0, a1);
#pragma unroll
      for (int off = 1; off < 16; off <<= 1) mx = fmaxf(mx, __shfl_xor(mx, off, 32));
      float mnew = fmaxf(mrow[p], mx);
      alpha[p] = __expf(mrow[p] - mnew);
      float e0 = __expf(a0 - mnew), e1 = __expf(a1 - mnew);
      Plds[wv][p + 8 * g][r]      = (_Float16)e0;
      Plds[wv][p + 8 * g][16 + r] = (_Float16)e1;
      float sum = e0 + e1;
#pragma unroll
      for (int off = 1; off < 16; off <<= 1) sum += __shfl_xor(sum, off, 32);
      lrow[p] = lrow[p] * alpha[p] + sum;
      mrow[p] = mnew;
    }
#pragma unroll
    for (int n = 0; n < 6; ++n)
#pragma unroll
      for (int p = 0; p < 8; ++p) o[n][p] *= alpha[p];

    asm volatile("s_wait_dscnt 0" ::: "memory");  // P stores visible before re-read

    // ---- P as A-fragment, then O += P @ V ----
    v16h pf;
#pragma unroll
    for (int e = 0; e < 16; ++e) pf[e] = Plds[wv][r][16 * g + e];
#pragma unroll
    for (int n = 0; n < 6; ++n) {
      const v16h vf = *(const v16h*)&Vt[n * 16 + r][16 * g];
      o[n] = WMMA_F16(pf, vf, o[n]);
    }
    __syncthreads();  // all waves done with K/Vt before restage
  }

  const size_t obase = ((size_t)(b * 8 + h) * 1024 + qt) * 96;
#pragma unroll
  for (int n = 0; n < 6; ++n)
#pragma unroll
    for (int p = 0; p < 8; ++p) {
      float val = o[n][p] / lrow[p];
      out[obase + (size_t)(p + 8 * g) * 96 + n * 16 + r] = (_Float16)val;
    }
}

// ---------------------------------------------------------------------------
// Orchestration
// ---------------------------------------------------------------------------
extern "C" void kernel_launch(void* const* d_in, const int* in_sizes, int n_in,
                              void* d_out, int out_size, void* d_ws, size_t ws_size,
                              hipStream_t stream) {
  (void)in_sizes; (void)n_in; (void)out_size; (void)ws_size;
  const float* x    = (const float*)d_in[0];
  const float* wqkv = (const float*)d_in[1];
  const float* wo   = (const float*)d_in[2];
  const float* bo   = (const float*)d_in[3];
  const float* w1   = (const float*)d_in[4];
  const float* b1   = (const float*)d_in[5];
  const float* w2   = (const float*)d_in[6];
  const float* b2   = (const float*)d_in[7];
  const float* g1   = (const float*)d_in[8];
  const float* be1  = (const float*)d_in[9];
  const float* gm   = (const float*)d_in[10];
  const float* bm   = (const float*)d_in[11];
  const float* g3   = (const float*)d_in[12];
  const float* be3  = (const float*)d_in[13];
  float* outp = (float*)d_out;

  // Workspace layout (bytes, all 256-aligned). a116 aliases [qkv16 + o16].
  char* ws = (char*)d_ws;
  _Float16* h16   = (_Float16*)(ws + 0);           // 8192x768  f16 (reused as h216)
  _Float16* qkv16 = (_Float16*)(ws + 12582912);    // 8192x2304 f16
  _Float16* o16   = (_Float16*)(ws + 50331648);    // 8192x768  f16
  _Float16* a116  = qkv16;                         // 8192x3072 f16 (aliases qkv16+o16)
  _Float16* wq16  = (_Float16*)(ws + 62914560);    // 768x2304  f16
  _Float16* wo16  = (_Float16*)(ws + 66453504);    // 768x768   f16
  _Float16* w116  = (_Float16*)(ws + 67633152);    // 768x3072  f16
  _Float16* w216  = (_Float16*)(ws + 72351744);    // 3072x768  f16
  float*    x1    = (float*)(ws + 77070336);       // 8192x768  f32 (post-attn residual)
  float*    x2    = (float*)(ws + 102236160);      // 8192x768  f32 (post-MLP)
  _Float16* h216  = h16;

  // 1) weights -> f16
  cast_f32_f16<<<4096, 256, 0, stream>>>(wqkv, wq16, 768 * 2304);
  cast_f32_f16<<<2048, 256, 0, stream>>>(wo,   wo16, 768 * 768);
  cast_f32_f16<<<4096, 256, 0, stream>>>(w1,   w116, 768 * 3072);
  cast_f32_f16<<<4096, 256, 0, stream>>>(w2,   w216, 3072 * 768);

  // 2) pre-norm 1 -> f16
  ln_kernel<true><<<8192, 256, 0, stream>>>(x, g1, be1, h16);

  // 3) QKV GEMM: [8192,768] x [768,2304] -> f16
  gemm_wmma<false, false, false, true>
      <<<dim3(2304 / 128, 8192 / 128), 256, 0, stream>>>(h16, wq16, nullptr, nullptr,
                                                         qkv16, 8192, 768, 2304);

  // 4) attention (head-major output == faithful reshape)
  attn_kernel<<<dim3(64, 16), 128, 0, stream>>>(qkv16, o16);

  // 5) output proj + b_o + residual(x): -> x1 f32
  gemm_wmma<true, false, true, false>
      <<<dim3(768 / 128, 8192 / 128), 256, 0, stream>>>(o16, wo16, bo, x, x1,
                                                        8192, 768, 768);

  // 6) pre-norm 2 -> f16
  ln_kernel<true><<<8192, 256, 0, stream>>>(x1, gm, bm, h216);

  // 7) MLP up: [8192,768] x [768,3072] + b1, GELU -> f16
  gemm_wmma<true, true, false, true>
      <<<dim3(3072 / 128, 8192 / 128), 256, 0, stream>>>(h216, w116, b1, nullptr,
                                                         a116, 8192, 768, 3072);

  // 8) MLP down: [8192,3072] x [3072,768] + b2, GELU, + residual(x1) -> x2 f32
  gemm_wmma<true, true, true, false>
      <<<dim3(768 / 128, 8192 / 128), 256, 0, stream>>>(a116, w216, b2, x1, x2,
                                                        8192, 3072, 768);

  // 9) final LayerNorm -> d_out f32
  ln_kernel<false><<<8192, 256, 0, stream>>>(x2, g3, be3, outp);
}
